// PointNetPlusPlus_52003464020760
// MI455X (gfx1250) — matrix operand order repaired and checked by reading, not compiled, not run on hardware
//
#include <hip/hip_runtime.h>
#include <hip/hip_bf16.h>
#include <float.h>

typedef __attribute__((ext_vector_type(16))) __bf16 v16bf;
typedef __attribute__((ext_vector_type(8)))  float  v8f;

#define BN_EPS 1e-5f
#define AS3 __attribute__((address_space(3)))

// ---------------------------------------------------------------------------
// Farthest point sampling: one block per batch; per-thread dist state in VGPRs,
// argmax via LDS tree reduction. Emits centroid index sequence + coords.
// ---------------------------------------------------------------------------
__global__ __launch_bounds__(256) void fps_kernel(const float* __restrict__ xyz,
                                                  int* __restrict__ cent,
                                                  float* __restrict__ new_xyz,
                                                  int N, int npoint)
{
    const int b   = blockIdx.x;
    const int tid = threadIdx.x;
    const float* P = xyz + (size_t)b * N * 3;

    __shared__ float s_val[256];
    __shared__ int   s_idx[256];
    __shared__ float s_c[3];

    float dists[32];                     // N <= 8192 -> <=32 points/thread
    const int cnt = (N + 255) >> 8;
    for (int j = 0; j < cnt; ++j) dists[j] = 1e10f;

    int far = 0;
    for (int it = 0; it < npoint; ++it) {
        if (tid == 0) {
            cent[b * npoint + it] = far;
            float cx = P[far * 3 + 0], cy = P[far * 3 + 1], cz = P[far * 3 + 2];
            s_c[0] = cx; s_c[1] = cy; s_c[2] = cz;
            new_xyz[((size_t)b * npoint + it) * 3 + 0] = cx;
            new_xyz[((size_t)b * npoint + it) * 3 + 1] = cy;
            new_xyz[((size_t)b * npoint + it) * 3 + 2] = cz;
        }
        __syncthreads();
        const float cx = s_c[0], cy = s_c[1], cz = s_c[2];

        float best = -1.0f; int bidx = 0;
        for (int j = 0; j < cnt; ++j) {
            int p = tid + (j << 8);
            if (p < N) {
                float dx = P[p * 3 + 0] - cx;
                float dy = P[p * 3 + 1] - cy;
                float dz = P[p * 3 + 2] - cz;
                float d  = dx * dx + dy * dy + dz * dz;
                float m  = fminf(dists[j], d);
                dists[j] = m;
                if (m > best) { best = m; bidx = p; }
            }
        }
        s_val[tid] = best; s_idx[tid] = bidx;
        __syncthreads();
        for (int s = 128; s > 0; s >>= 1) {
            if (tid < s) {
                float v2 = s_val[tid + s]; int i2 = s_idx[tid + s];
                if (v2 > s_val[tid] || (v2 == s_val[tid] && i2 < s_idx[tid])) {
                    s_val[tid] = v2; s_idx[tid] = i2;
                }
            }
            __syncthreads();
        }
        far = s_idx[0];
        __syncthreads();
    }
}

// ---------------------------------------------------------------------------
// Ball query (top-K nearest, radius-clamped to nearest): one block per center.
// All N squared distances cached in LDS; K iterative argmin extractions.
// ---------------------------------------------------------------------------
__global__ __launch_bounds__(256) void ball_query_kernel(const float* __restrict__ xyz,
                                                         const float* __restrict__ nxyz,
                                                         int* __restrict__ idx,
                                                         int S, int N, int K, float r2)
{
    const int bs  = blockIdx.x;
    const int b   = bs / S;
    const int tid = threadIdx.x;

    __shared__ float sd[8192];
    __shared__ float s_val[256];
    __shared__ int   s_idx[256];
    __shared__ int   s_win0;

    const float cx = nxyz[bs * 3 + 0];
    const float cy = nxyz[bs * 3 + 1];
    const float cz = nxyz[bs * 3 + 2];
    const float* P = xyz + (size_t)b * N * 3;

    for (int p = tid; p < N; p += 256) {
        float dx = P[p * 3 + 0] - cx;
        float dy = P[p * 3 + 1] - cy;
        float dz = P[p * 3 + 2] - cz;
        sd[p] = dx * dx + dy * dy + dz * dz;
    }
    __syncthreads();

    for (int k = 0; k < K; ++k) {
        float best = FLT_MAX; int bidx = 0x7fffffff;
        for (int p = tid; p < N; p += 256) {
            float v = sd[p];
            if (v < best || (v == best && p < bidx)) { best = v; bidx = p; }
        }
        s_val[tid] = best; s_idx[tid] = bidx;
        __syncthreads();
        for (int t = 128; t > 0; t >>= 1) {
            if (tid < t) {
                float v2 = s_val[tid + t]; int i2 = s_idx[tid + t];
                if (v2 < s_val[tid] || (v2 == s_val[tid] && i2 < s_idx[tid])) {
                    s_val[tid] = v2; s_idx[tid] = i2;
                }
            }
            __syncthreads();
        }
        int win = s_idx[0]; float wv = s_val[0];
        if (k == 0 && tid == 0) s_win0 = win;
        __syncthreads();
        if (tid == 0) {
            idx[(size_t)bs * K + k] = (wv > r2) ? s_win0 : win;
            sd[win] = FLT_MAX;
        }
        __syncthreads();
    }
}

// ---------------------------------------------------------------------------
// Gather grouped features (xyz - center, concat point features), write bf16
// GEMM input with K padded to multiple of 32 (zeros).
// ---------------------------------------------------------------------------
__global__ void build_group_kernel(const float* __restrict__ xyz,
                                   const float* __restrict__ nxyz,
                                   const float* __restrict__ points,
                                   const int* __restrict__ idx,
                                   __bf16* __restrict__ X,
                                   int S, int N, int K, int Cpts, int Kpad,
                                   long long total)
{
    long long gid = (long long)blockIdx.x * blockDim.x + threadIdx.x;
    if (gid >= total) return;
    int c = (int)(gid % Kpad);
    long long r = gid / Kpad;
    int b   = (int)(r / ((long long)S * K));
    int rem = (int)(r % ((long long)S * K));
    int s = rem / K;
    int k = rem % K;
    int i = idx[((size_t)b * S + s) * K + k];

    float v = 0.0f;
    if (c < 3) {
        v = xyz[((size_t)b * N + i) * 3 + c] - nxyz[((size_t)b * S + s) * 3 + c];
    } else if (c < 3 + Cpts) {
        v = points[((size_t)b * N + i) * Cpts + (c - 3)];
    }
    X[gid] = (__bf16)v;
}

// ---------------------------------------------------------------------------
// Per-layer prep: W f32 [cout,cin] -> bf16 [cout,Kpad]; fold bias + eval-BN
// into per-channel scale/shift:  y = scale*x + shift.
// ---------------------------------------------------------------------------
__global__ void prep_weights_kernel(const float* __restrict__ W,
                                    const float* __restrict__ bias,
                                    const float* __restrict__ gamma,
                                    const float* __restrict__ beta,
                                    const float* __restrict__ mean,
                                    const float* __restrict__ var,
                                    __bf16* __restrict__ Wb,
                                    float* __restrict__ scale,
                                    float* __restrict__ shift,
                                    int cout, int cin, int Kpad)
{
    int gid = blockIdx.x * blockDim.x + threadIdx.x;
    int total = cout * Kpad;
    if (gid < total) {
        int n = gid / Kpad, k = gid % Kpad;
        Wb[gid] = (k < cin) ? (__bf16)W[(size_t)n * cin + k] : (__bf16)0.0f;
    }
    if (gid < cout) {
        float sc = gamma[gid] * rsqrtf(var[gid] + BN_EPS);
        scale[gid] = sc;
        shift[gid] = beta[gid] + sc * (bias[gid] - mean[gid]);
    }
}

// ---------------------------------------------------------------------------
// WMMA GEMM:  Y[rows,cout] = relu(scale * (X[rows,K] @ Wb[cout,K]^T) + shift)
// Block = 8 waves -> 128 rows x 64 cols. Each wave: one 16-row tile x 4
// column tiles (A fragment reused 4x -> 4x arithmetic intensity on the
// activation stream). The shared 64x32 bf16 weight panel is double-buffered
// in LDS via GLOBAL_LOAD_ASYNC_TO_LDS_B128 (ASYNCcnt), overlapped with WMMA.
// Fragment layouts per CDNA5 ISA 7.12.2 (wave32).
// ---------------------------------------------------------------------------
__global__ __launch_bounds__(256) void gemm_bn_relu_kernel(
    const __bf16* __restrict__ X, const __bf16* __restrict__ Wb,
    const float* __restrict__ scale, const float* __restrict__ shift,
    __bf16* __restrict__ Y, int rows, int K, int cout)
{
    __shared__ __bf16 sB[2][64 * 32];     // double-buffered weight panel (8KB)

    const int wave = threadIdx.x >> 5;
    const int lane = threadIdx.x & 31;
    const int rowTiles = rows >> 4;
    int rowTile = blockIdx.x * 8 + wave;
    if (rowTile >= rowTiles) rowTile %= rowTiles;  // keep all waves cooperating
    const int n0 = blockIdx.y * 64;
    const int m  = lane & 15;             // A row / B col within tile
    const int h  = lane >> 4;             // lane half

    const unsigned int* Xu = (const unsigned int*)(X + (size_t)(rowTile * 16 + m) * K);

    // cooperative async panel loader: 256 threads x 16B = 4KB per K-step
    const int prow = threadIdx.x >> 2;    // 0..63: panel row (output channel)
    const int pq   = threadIdx.x & 3;     // 0..3: 16B chunk within 64B row
    const __bf16* wsrc0 = Wb + (size_t)(n0 + prow) * K + pq * 8;
    unsigned ldsDst0 = (unsigned)(unsigned long long)(AS3 const char*)
                           &sB[0][prow * 32 + pq * 8];
    unsigned ldsDst1 = (unsigned)(unsigned long long)(AS3 const char*)
                           &sB[1][prow * 32 + pq * 8];

    auto loadB = [&](int k0, int buf) {
        unsigned long long ga = (unsigned long long)(wsrc0 + k0);
        unsigned ld = buf ? ldsDst1 : ldsDst0;
        asm volatile("global_load_async_to_lds_b128 %0, %1, off"
                     :: "v"(ld), "v"(ga) : "memory");
    };

    v8f acc[4] = {};

    loadB(0, 0);
    asm volatile("s_wait_asynccnt 0x0" ::: "memory");
    __syncthreads();

    int cur = 0;
    for (int k0 = 0; k0 < K; k0 += 32) {
        if (k0 + 32 < K) {
            loadB(k0 + 32, cur ^ 1);                       // prefetch weights
            __builtin_prefetch(Xu + ((k0 + 32) >> 1), 0, 0);  // prefetch A
        }

        // A 16x32 bf16: VGPR q -> K = (q>=4?16:0) + 8*h + 2*(q&3) (+0,+1)
        union { v16bf v; unsigned int u[8]; } A;
        #pragma unroll
        for (int q = 0; q < 8; ++q) {
            int kk = ((q >= 4) ? 16 : 0) + h * 8 + ((q & 3) << 1);
            A.u[q] = Xu[(k0 + kk) >> 1];
        }

        // 4 column tiles: B fragment lane n holds K = (e>=8?16:0) + 8*h + (e&7)
        #pragma unroll
        for (int t = 0; t < 4; ++t) {
            union { v16bf v; unsigned int u[8]; } Bf;
            const unsigned int* bp =
                (const unsigned int*)&sB[cur][(t * 16 + m) * 32 + h * 8];
            #pragma unroll
            for (int q = 0; q < 8; ++q) Bf.u[q] = bp[(q & 3) + ((q >> 2) << 3)];
            acc[t] = __builtin_amdgcn_wmma_f32_16x16x32_bf16(
                false, A.v, false, Bf.v, (short)0, acc[t], false, false);
        }

        asm volatile("s_wait_asynccnt 0x0" ::: "memory");
        __syncthreads();
        cur ^= 1;
    }

    // C/D layout: VGPR j -> M = 8*h + j, N = lane&15
    #pragma unroll
    for (int t = 0; t < 4; ++t) {
        int nn = n0 + t * 16 + m;
        float sc = scale[nn], sh = shift[nn];
        #pragma unroll
        for (int j = 0; j < 8; ++j) {
            int mm = h * 8 + j;
            float y = fmaxf(acc[t][j] * sc + sh, 0.0f);
            Y[(size_t)(rowTile * 16 + mm) * cout + nn] = (__bf16)y;
        }
    }
}

// ---------------------------------------------------------------------------
// Max-pool over the group (K) dimension: bf16 activations -> f32 pooled.
// ---------------------------------------------------------------------------
__global__ void maxpool_kernel(const __bf16* __restrict__ Yin,
                               float* __restrict__ pooled,
                               int K, int C, long long total)
{
    long long gid = (long long)blockIdx.x * blockDim.x + threadIdx.x;
    if (gid >= total) return;
    long long g = gid / C;
    int c = (int)(gid % C);
    float mx = -FLT_MAX;
    for (int k = 0; k < K; ++k) {
        float v = (float)Yin[((size_t)(g * K + k)) * C + c];
        mx = fmaxf(mx, v);
    }
    pooled[gid] = mx;
}

// ---------------------------------------------------------------------------
// Build regression-head input: concat(pooled3[1024], load_dir[3], phys[13]),
// zero-pad to Kpad, bf16.
// ---------------------------------------------------------------------------
__global__ void build_reg_kernel(const float* __restrict__ pooled3,
                                 const float* __restrict__ load_dir,
                                 const float* __restrict__ phys,
                                 __bf16* __restrict__ X, int Kpad, int total)
{
    int gid = blockIdx.x * blockDim.x + threadIdx.x;
    if (gid >= total) return;
    int b = gid / Kpad, c = gid % Kpad;
    float v = 0.0f;
    if (c < 1024)      v = pooled3[b * 1024 + c];
    else if (c < 1027) v = load_dir[b * 3 + (c - 1024)];
    else if (c < 1040) v = phys[b * 13 + (c - 1027)];
    X[gid] = (__bf16)v;
}

// Final [16,128] @ [128,1] + b
__global__ void final_dot_kernel(const __bf16* __restrict__ X,
                                 const float* __restrict__ W,
                                 const float* __restrict__ b,
                                 float* __restrict__ out, int Bn, int C)
{
    int i = threadIdx.x;
    if (i >= Bn) return;
    float s = 0.0f;
    for (int c = 0; c < C; ++c) s += (float)X[(size_t)i * C + c] * W[c];
    out[i] = s + b[0];
}

// ---------------------------------------------------------------------------
// Host orchestration
// ---------------------------------------------------------------------------
static inline int pad32i(int x) { return (x + 31) & ~31; }

extern "C" void kernel_launch(void* const* d_in, const int* in_sizes, int n_in,
                              void* d_out, int out_size, void* d_ws, size_t ws_size,
                              hipStream_t stream)
{
    (void)in_sizes; (void)n_in; (void)out_size; (void)ws_size;

    const float* xyz      = (const float*)d_in[0];   // [16,8192,3]
    const float* load_dir = (const float*)d_in[1];   // [16,3]
    const float* phys     = (const float*)d_in[2];   // [16,13]

    // params flattened in insertion order: sa1,sa2,sa3 (3 layers each),
    // reg (3 BN layers + final {W,b}); each BN layer: W,b,gamma,beta,mean,var
    auto LP = [&](int layer, int field) -> const float* {
        return (const float*)d_in[3 + layer * 6 + field];
    };
    const float* regWf = (const float*)d_in[3 + 12 * 6 + 0]; // [1,128]
    const float* regbf = (const float*)d_in[3 + 12 * 6 + 1]; // [1]

    // layer dims (cin, cout), layers 0..11
    static const int CIN[12]  = {3, 64, 64,   131, 128, 128,  259, 256, 512,  1040, 512, 256};
    static const int COUT[12] = {64, 64, 128, 128, 128, 256,  256, 512, 1024, 512,  256, 128};

    // ---- bump allocator over d_ws ----
    char* ws = (char*)d_ws;
    size_t off = 0;
    auto alloc = [&](size_t bytes) -> void* {
        off = (off + 255) & ~(size_t)255;
        void* p = ws + off;
        off += bytes;
        return p;
    };

    const int B = 16;
    int*   cent1 = (int*)alloc((size_t)B * 512 * 4);
    float* nx1   = (float*)alloc((size_t)B * 512 * 3 * 4);
    int*   idx1  = (int*)alloc((size_t)B * 512 * 32 * 4);
    int*   cent2 = (int*)alloc((size_t)B * 128 * 4);
    float* nx2   = (float*)alloc((size_t)B * 128 * 3 * 4);
    int*   idx2  = (int*)alloc((size_t)B * 128 * 64 * 4);
    int*   cent3 = (int*)alloc((size_t)B * 1 * 4);
    float* nx3   = (float*)alloc((size_t)B * 1 * 3 * 4);
    int*   idx3  = (int*)alloc((size_t)B * 1 * 128 * 4);
    float* pooled1 = (float*)alloc((size_t)B * 512 * 128 * 4);
    float* pooled2 = (float*)alloc((size_t)B * 128 * 256 * 4);
    float* pooled3 = (float*)alloc((size_t)B * 1024 * 4);
    __bf16* wb    = (__bf16*)alloc((size_t)1024 * 1056 * 2);   // max cout*Kpad
    float*  scale = (float*)alloc(1024 * 4);
    float*  shift = (float*)alloc(1024 * 4);
    const size_t ACT_ELEMS = 33554432ull;                      // 262144*128
    __bf16* ping = (__bf16*)alloc(ACT_ELEMS * 2);
    __bf16* pong = (__bf16*)alloc(ACT_ELEMS * 2);

    auto run_layer = [&](int layer, const __bf16* Xin, __bf16* Yout, int rows) {
        int cin = CIN[layer], cout = COUT[layer], Kp = pad32i(cin);
        int wtot = cout * Kp;
        prep_weights_kernel<<<(wtot + 255) / 256, 256, 0, stream>>>(
            LP(layer, 0), LP(layer, 1), LP(layer, 2), LP(layer, 3),
            LP(layer, 4), LP(layer, 5), wb, scale, shift, cout, cin, Kp);
        dim3 grid((rows / 16 + 7) / 8, cout / 64);   // 8 row tiles, 64 cols/block
        gemm_bn_relu_kernel<<<grid, 256, 0, stream>>>(
            Xin, wb, scale, shift, Yout, rows, Kp, cout);
    };

    // ================= SA1: N=8192 -> S=512, K=32, r=0.2 =================
    {
        const int N = 8192, S = 512, K = 32;
        fps_kernel<<<B, 256, 0, stream>>>(xyz, cent1, nx1, N, S);
        ball_query_kernel<<<B * S, 256, 0, stream>>>(xyz, nx1, idx1, S, N, K, 0.04f);
        const int Kp0 = pad32i(3);
        long long rows = (long long)B * S * K;            // 262144
        long long tot = rows * Kp0;
        build_group_kernel<<<(int)((tot + 255) / 256), 256, 0, stream>>>(
            xyz, nx1, nullptr, idx1, ping, S, N, K, 0, Kp0, tot);
        run_layer(0, ping, pong, (int)rows);              // 3->64
        run_layer(1, pong, ping, (int)rows);              // 64->64
        run_layer(2, ping, pong, (int)rows);              // 64->128
        long long ptot = (long long)B * S * 128;
        maxpool_kernel<<<(int)((ptot + 255) / 256), 256, 0, stream>>>(
            pong, pooled1, K, 128, ptot);
    }

    // ================= SA2: N=512 -> S=128, K=64, r=0.4 =================
    {
        const int N = 512, S = 128, K = 64;
        fps_kernel<<<B, 256, 0, stream>>>(nx1, cent2, nx2, N, S);
        ball_query_kernel<<<B * S, 256, 0, stream>>>(nx1, nx2, idx2, S, N, K, 0.16f);
        const int Kp0 = pad32i(131);                      // 160
        long long rows = (long long)B * S * K;            // 131072
        long long tot = rows * Kp0;
        build_group_kernel<<<(int)((tot + 255) / 256), 256, 0, stream>>>(
            nx1, nx2, pooled1, idx2, ping, S, N, K, 128, Kp0, tot);
        run_layer(3, ping, pong, (int)rows);              // 131->128
        run_layer(4, pong, ping, (int)rows);              // 128->128
        run_layer(5, ping, pong, (int)rows);              // 128->256
        long long ptot = (long long)B * S * 256;
        maxpool_kernel<<<(int)((ptot + 255) / 256), 256, 0, stream>>>(
            pong, pooled2, K, 256, ptot);
    }

    // ================= SA3: N=128 -> S=1, K=128, r=2.0 =================
    {
        const int N = 128, S = 1, K = 128;
        fps_kernel<<<B, 256, 0, stream>>>(nx2, cent3, nx3, N, S);
        ball_query_kernel<<<B * S, 256, 0, stream>>>(nx2, nx3, idx3, S, N, K, 4.0f);
        const int Kp0 = pad32i(259);                      // 288
        long long rows = (long long)B * S * K;            // 2048
        long long tot = rows * Kp0;
        build_group_kernel<<<(int)((tot + 255) / 256), 256, 0, stream>>>(
            nx2, nx3, pooled2, idx3, ping, S, N, K, 256, Kp0, tot);
        run_layer(6, ping, pong, (int)rows);              // 259->256
        run_layer(7, pong, ping, (int)rows);              // 256->512
        run_layer(8, ping, pong, (int)rows);              // 512->1024
        long long ptot = (long long)B * 1024;
        maxpool_kernel<<<(int)((ptot + 255) / 256), 256, 0, stream>>>(
            pong, pooled3, K, 1024, ptot);
    }

    // ================= regression head =================
    {
        const int Kp0 = pad32i(1040);                     // 1056
        int tot = B * Kp0;
        build_reg_kernel<<<(tot + 255) / 256, 256, 0, stream>>>(
            pooled3, load_dir, phys, ping, Kp0, tot);
        run_layer(9,  ping, pong, B);                     // 1040->512
        run_layer(10, pong, ping, B);                     // 512->256
        run_layer(11, ping, pong, B);                     // 256->128
        final_dot_kernel<<<1, 32, 0, stream>>>(pong, regWf, regbf,
                                               (float*)d_out, B, 128);
    }
}
